// CausalSelfAttention_30975304139002
// MI455X (gfx1250) — compile-verified, matrix operand order checked by reading, not run on hardware
//
#include <hip/hip_runtime.h>

typedef unsigned short ushort_t;
typedef __attribute__((ext_vector_type(16))) __bf16 bf16x16;
typedef __attribute__((ext_vector_type(8)))  float  f32x8;

struct U4 { unsigned int x, y, z, w; };
union FragU { bf16x16 v; U4 q[2]; };

__device__ __forceinline__ unsigned short f2bf(float f) {
  unsigned int u = __float_as_uint(f);
  u += 0x7FFFu + ((u >> 16) & 1u);           // round-to-nearest-even
  return (unsigned short)(u >> 16);
}

__device__ __forceinline__ f32x8 zero8() {
  f32x8 z; for (int e = 0; e < 8; ++e) z[e] = 0.f; return z;
}

// CDNA5 async DMA: global -> LDS, 16B per lane, tracked by ASYNCcnt.
__device__ __forceinline__ void async_b128(ushort_t* lds_dst, const ushort_t* g_src) {
  unsigned l = (unsigned)(uintptr_t)lds_dst;  // low 32b of shared generic ptr = LDS offset
  asm volatile("global_load_async_to_lds_b128 %0, %1, off"
               :: "v"(l), "v"(g_src)
               : "memory");
}
__device__ __forceinline__ void wait_async0() {
#if __has_builtin(__builtin_amdgcn_s_wait_asynccnt)
  __builtin_amdgcn_s_wait_asynccnt(0);
#else
  asm volatile("s_wait_asynccnt 0x0" ::: "memory");
#endif
}

// Load a 16x32 bf16 A/B fragment from a row-major [rows][ld] bf16 matrix.
// CDNA5 16-bit layout: lane m=lane&15; lanes 0-15 hold K {0..7,16..23},
// lanes 16-31 hold K {8..15,24..31} (pairs per VGPR) -> two 16B reads.
__device__ __forceinline__ bf16x16 load_frag(const ushort_t* base, int ld, int row, int kbase) {
  const int lane = threadIdx.x & 31;
  const int idx  = lane & 15;
  const int kof  = (lane >> 4) << 3;
  const ushort_t* p = base + (size_t)(row + idx) * ld + kbase + kof;
  FragU f;
  f.q[0] = *reinterpret_cast<const U4*>(p);
  f.q[1] = *reinterpret_cast<const U4*>(p + 16);
  return f.v;
}

__device__ __forceinline__ f32x8 wmma_bf16(bf16x16 a, bf16x16 b, f32x8 c) {
  return __builtin_amdgcn_wmma_f32_16x16x32_bf16(false, a, false, b, (short)0, c, false, false);
}

// ---------------- converters ----------------
__global__ __launch_bounds__(256) void cvt_bf16_kernel(const float* __restrict__ in,
                                                       ushort_t* __restrict__ out, int n8) {
  int i = blockIdx.x * blockDim.x + threadIdx.x;
  if (i >= n8) return;
  const float* p = in + (size_t)i * 8;
  U4 o;
  o.x = f2bf(p[0]) | ((unsigned)f2bf(p[1]) << 16);
  o.y = f2bf(p[2]) | ((unsigned)f2bf(p[3]) << 16);
  o.z = f2bf(p[4]) | ((unsigned)f2bf(p[5]) << 16);
  o.w = f2bf(p[6]) | ((unsigned)f2bf(p[7]) << 16);
  *reinterpret_cast<U4*>(out + (size_t)i * 8) = o;
}

// W is [K][N] fp32; write Wt as [N][K] bf16 (coalesced on output).
__global__ __launch_bounds__(256) void cvt_t_kernel(const float* __restrict__ in,
                                                    ushort_t* __restrict__ out, int K, int N) {
  size_t idx = (size_t)blockIdx.x * blockDim.x + threadIdx.x;
  if (idx >= (size_t)K * N) return;
  int n  = (int)(idx / K);
  int kk = (int)(idx % K);
  out[idx] = f2bf(in[(size_t)kk * N + n]);
}

// ---------------- tiled bf16 WMMA GEMM: C[M][N] = A[M][K] * Bt[N][K]^T -------------
#define BM 128
#define BN 128
#define BK 32
#define LDSW 40   // padded bf16 stride: 80B rows, 16B-aligned fragment reads

template <int MODE>  // 0: fp32 C row-major; 1: scatter into q/k/v^T bf16
__global__ __launch_bounds__(256) void gemm_bf16_wmma(
    const ushort_t* __restrict__ A, const ushort_t* __restrict__ Bt,
    float* __restrict__ Cf,
    ushort_t* __restrict__ qo, ushort_t* __restrict__ ko, ushort_t* __restrict__ vto,
    int M, int N, int K) {
  __shared__ ushort_t As[2 * BM * LDSW];
  __shared__ ushort_t Bs[2 * BN * LDSW];
  const int tid  = threadIdx.x;
  const int lane = tid & 31;
  const int wid  = tid >> 5;
  const int wm   = wid & 1;   // 2 waves along M (64 rows each)
  const int wn   = wid >> 1;  // 4 waves along N (32 cols each)
  const int bm = blockIdx.y, bn = blockIdx.x;
  const int half = lane >> 4, nl = lane & 15;

  const int lrow = tid >> 2;         // 0..63
  const int lcol = (tid & 3) << 3;   // 0,8,16,24  (8 bf16 = 16B)
  const size_t arow0 = (size_t)bm * BM + lrow;
  const size_t brow0 = (size_t)bn * BN + lrow;

  // Async DMA a full 128x32 A tile + 128x32 B tile straight into LDS.
  auto issue_tile = [&](int buf, int k0) {
    async_b128(&As[(buf * BM + lrow     ) * LDSW + lcol], A  + (arow0      ) * K + k0 + lcol);
    async_b128(&As[(buf * BM + lrow + 64) * LDSW + lcol], A  + (arow0 + 64 ) * K + k0 + lcol);
    async_b128(&Bs[(buf * BN + lrow     ) * LDSW + lcol], Bt + (brow0      ) * K + k0 + lcol);
    async_b128(&Bs[(buf * BN + lrow + 64) * LDSW + lcol], Bt + (brow0 + 64 ) * K + k0 + lcol);
  };

  f32x8 acc[4][2];
  for (int i = 0; i < 4; ++i) for (int j = 0; j < 2; ++j) acc[i][j] = zero8();

  issue_tile(0, 0);
  wait_async0();
  __syncthreads();
  const int nsteps = K / BK;
  for (int s = 0; s < nsteps; ++s) {
    const int buf = s & 1;
    if (s + 1 < nsteps) issue_tile(buf ^ 1, (s + 1) * BK);   // DMA overlaps WMMAs below
    if (s + 2 < nsteps) {
      __builtin_prefetch(A  + arow0 * K + (s + 2) * BK + lcol, 0, 1);
      __builtin_prefetch(Bt + brow0 * K + (s + 2) * BK + lcol, 0, 1);
    }
    const ushort_t* as = &As[buf * BM * LDSW];
    const ushort_t* bs = &Bs[buf * BN * LDSW];
    bf16x16 af[4], bfr[2];
    for (int mt = 0; mt < 4; ++mt) af[mt]  = load_frag(as, LDSW, wm * 64 + mt * 16, 0);
    for (int nt = 0; nt < 2; ++nt) bfr[nt] = load_frag(bs, LDSW, wn * 32 + nt * 16, 0);
    for (int mt = 0; mt < 4; ++mt)
      for (int nt = 0; nt < 2; ++nt)
        acc[mt][nt] = wmma_bf16(af[mt], bfr[nt], acc[mt][nt]);
    if (s + 1 < nsteps) wait_async0();
    __syncthreads();
  }

  // Epilogue: C layout lane->(N=nl, M=i+8*half) per accumulator VGPR i.
  for (int mt = 0; mt < 4; ++mt)
    for (int nt = 0; nt < 2; ++nt)
      for (int i = 0; i < 8; ++i) {
        const int r    = bm * BM + wm * 64 + mt * 16 + i + half * 8;
        const int cidx = bn * BN + wn * 32 + nt * 16 + nl;
        const float val = acc[mt][nt][i];
        if (MODE == 0) {
          Cf[(size_t)r * N + cidx] = val;
        } else {
          const int sec = cidx / 768;
          const int c   = cidx - sec * 768;
          const int h = c >> 6, d = c & 63;
          const int b = r >> 11, t = r & 2047;
          const size_t bh = (size_t)(b * 12 + h);
          const unsigned short bv = f2bf(val);
          if (sec == 0)      qo [(bh * 2048 + t) * 64 + d] = bv;
          else if (sec == 1) ko [(bh * 2048 + t) * 64 + d] = bv;
          else               vto[(bh * 64 + d) * 2048 + t] = bv;  // V transposed for PV wmma
        }
      }
}

// ---------------- fused causal ReLU attention (flash-style, no softmax rescale) ----
#define ALD 72  // padded bf16 stride: 144B rows

__global__ __launch_bounds__(128) void relu_attn_wmma(
    const ushort_t* __restrict__ q, const ushort_t* __restrict__ k,
    const ushort_t* __restrict__ vt,
    const float* __restrict__ amask, const float* __restrict__ lam,
    ushort_t* __restrict__ aout) {
  __shared__ ushort_t Ks[2 * 64 * ALD];
  __shared__ ushort_t Vs[2 * 64 * ALD];
  __shared__ ushort_t Ss[4 * 16 * ALD];  // per-wave score staging (C->A re-stripe)
  const int tid = threadIdx.x, lane = tid & 31, w = tid >> 5;
  const int half = lane >> 4, nl = lane & 15;
  const int bid = blockIdx.x;
  const int qt  = bid & 31;       // q tile (64 rows) within T
  const int bh  = bid >> 5;       // (b,h)
  const int h = bh % 12, b = bh / 12;
  const size_t bhoff = (size_t)bh * 2048 * 64;
  const ushort_t* Qp = q  + bhoff;
  const ushort_t* Kp = k  + bhoff;
  const ushort_t* Vp = vt + bhoff;

  const int qrow0 = qt * 64 + w * 16;
  bf16x16 Qf[2];
  Qf[0] = load_frag(Qp, 64, qrow0, 0);
  Qf[1] = load_frag(Qp, 64, qrow0, 32);

  const float lamh = lam[h];
  float rowscale[8], rowsum[8];
  for (int i = 0; i < 8; ++i) {
    const int gq = qrow0 + i + half * 8;
    rowscale[i] = 0.125f * (1.f + lamh * __logf((float)(gq + 1)));  // 1/sqrt(64)=0.125
    rowsum[i] = 0.f;
  }
  f32x8 O[4];
  for (int nt = 0; nt < 4; ++nt) O[nt] = zero8();

  const int ldr = tid >> 3;         // 0..15
  const int ldc = (tid & 7) << 3;   // 0..56
  // Async DMA K-tile [key][d] and V^T-tile [d][key] into LDS.
  auto loadKV = [&](int buf, int kt) {
    for (int p = 0; p < 4; ++p) {
      const int row = p * 16 + ldr;
      async_b128(&Ks[(buf * 64 + row) * ALD + ldc],
                 Kp + (size_t)(kt * 64 + row) * 64 + ldc);
      async_b128(&Vs[(buf * 64 + row) * ALD + ldc],
                 Vp + (size_t)row * 2048 + kt * 64 + ldc);
    }
  };

  loadKV(0, 0);
  wait_async0();
  __syncthreads();
  for (int kt = 0; kt <= qt; ++kt) {
    const int buf = kt & 1;
    if (kt < qt) loadKV(buf ^ 1, kt + 1);   // DMA next tile, overlap with WMMAs
    const ushort_t* ks_ = &Ks[buf * 64 * ALD];
    const ushort_t* vs_ = &Vs[buf * 64 * ALD];

    // S = Q @ K^T over D=64 (two 16x16x32 steps per 16-key tile)
    f32x8 S[4];
    for (int nt = 0; nt < 4; ++nt) {
      f32x8 a = zero8();
      a = wmma_bf16(Qf[0], load_frag(ks_, ALD, nt * 16, 0),  a);
      a = wmma_bf16(Qf[1], load_frag(ks_, ALD, nt * 16, 32), a);
      S[nt] = a;
    }

    // scale * relu * (causal,mask); accumulate denominator; stage P as bf16
    ushort_t* ss = &Ss[w * 16 * ALD];
    for (int nt = 0; nt < 4; ++nt) {
      const int gk = kt * 64 + nt * 16 + nl;
      const float mv = amask[b * 2048 + gk];
      for (int i = 0; i < 8; ++i) {
        const int gq = qrow0 + i + half * 8;
        float v = S[nt][i] * rowscale[i];
        v = fmaxf(v, 0.f);
        v = (gk <= gq) ? v * mv : 0.f;
        rowsum[i] += v;
        ss[(i + half * 8) * ALD + nt * 16 + nl] = f2bf(v);
      }
    }

    // O += P @ V^T  (Vs is [d][key] so B fragments are contiguous)
    bf16x16 Pf0 = load_frag(ss, ALD, 0, 0);
    bf16x16 Pf1 = load_frag(ss, ALD, 0, 32);
    for (int nt = 0; nt < 4; ++nt) {
      O[nt] = wmma_bf16(Pf0, load_frag(vs_, ALD, nt * 16, 0),  O[nt]);
      O[nt] = wmma_bf16(Pf1, load_frag(vs_, ALD, nt * 16, 32), O[nt]);
    }
    if (kt < qt) wait_async0();
    __syncthreads();
  }

  // row-sum reduce across the 16 lanes of each half, then normalize + store
  for (int i = 0; i < 8; ++i) {
    float s = rowsum[i];
    s += __shfl_xor(s, 1, 32);
    s += __shfl_xor(s, 2, 32);
    s += __shfl_xor(s, 4, 32);
    s += __shfl_xor(s, 8, 32);
    rowsum[i] = s;
  }
  for (int nt = 0; nt < 4; ++nt)
    for (int i = 0; i < 8; ++i) {
      const int gq = qrow0 + i + half * 8;
      const float val = O[nt][i] / (rowsum[i] + 1e-9f);
      aout[((size_t)(b * 2048 + gq)) * 768 + h * 64 + nt * 16 + nl] = f2bf(val);
    }
}

// ---------------- launch ----------------
extern "C" void kernel_launch(void* const* d_in, const int* in_sizes, int n_in,
                              void* d_out, int out_size, void* d_ws, size_t ws_size,
                              hipStream_t stream) {
  const float* x  = (const float*)d_in[0];
  const float* am = (const float*)d_in[1];
  const float* lm = (const float*)d_in[2];
  const float* Wa = (const float*)d_in[3];
  const float* Wp = (const float*)d_in[4];
  float* out = (float*)d_out;

  char* w = (char*)d_ws;
  ushort_t* xbf = (ushort_t*)(w);              // x bf16        8192x768
  ushort_t* wat = (ushort_t*)(w + 12582912);   // W_attn^T bf16 2304x768
  ushort_t* wpt = (ushort_t*)(w + 16121856);   // W_proj^T bf16  768x768
  ushort_t* qb  = (ushort_t*)(w + 17301504);   // q  [b,h,t,d]
  ushort_t* kb  = (ushort_t*)(w + 29884416);   // k  [b,h,t,d]
  ushort_t* vtb = (ushort_t*)(w + 42467328);   // v^T[b,h,d,t]
  ushort_t* ab  = (ushort_t*)(w + 55050240);   // attn out bf16 8192x768
  (void)in_sizes; (void)n_in; (void)out_size; (void)ws_size;

  cvt_bf16_kernel<<<3072, 256, 0, stream>>>(x, xbf, 786432);          // 8192*768/8
  cvt_t_kernel<<<6912, 256, 0, stream>>>(Wa, wat, 768, 2304);
  cvt_t_kernel<<<2304, 256, 0, stream>>>(Wp, wpt, 768, 768);
  gemm_bf16_wmma<1><<<dim3(18, 64), 256, 0, stream>>>(xbf, wat, nullptr,
                                                      qb, kb, vtb, 8192, 2304, 768);
  relu_attn_wmma<<<1536, 128, 0, stream>>>(qb, kb, vtb, am, lm, ab);
  gemm_bf16_wmma<0><<<dim3(6, 64), 256, 0, stream>>>(ab, wpt, out,
                                                     nullptr, nullptr, nullptr, 8192, 768, 768);
}
